// MultiHeadSelfAttention_30863634989278
// MI455X (gfx1250) — matrix operand order repaired and checked by reading, not compiled, not run on hardware
//
#include <hip/hip_runtime.h>
#include <hip/hip_bf16.h>

// ---------------------------------------------------------------------------
// MHSA block for MI455X (gfx1250): bf16 WMMA (fp32 accumulate) for all four
// GEMM stages + flash-style online-softmax attention, fp32 LayerNorm epilogue.
// LDS tile staging is done by the Tensor Data Mover (tensor_load_to_lds).
// ---------------------------------------------------------------------------

typedef __attribute__((ext_vector_type(16))) __bf16 v16bf;
typedef __attribute__((ext_vector_type(8)))  float  v8f;
typedef unsigned int u32x4 __attribute__((ext_vector_type(4)));
typedef int          i32x4 __attribute__((ext_vector_type(4)));
typedef int          i32x8 __attribute__((ext_vector_type(8)));

#define EMBED    1024
#define HEADS    16
#define HEAD_DIM 64
#define SEQ      2048
#define BATCH    4
#define ROWS     (BATCH * SEQ)      // 8192

union BF16x16 {
    v16bf v;
    unsigned int u[8];              // u[i] = {bf16 lo, bf16 hi}
};

static __device__ __forceinline__ v8f wmma_bf16(v16bf a, v16bf b, v8f c) {
    // D = A(16x32) * B(32x16) + C(16x16 f32)
    return __builtin_amdgcn_wmma_f32_16x16x32_bf16(
        /*neg_a=*/false, a, /*neg_b=*/false, b,
        /*c_mod=*/(short)0, c, /*reuse_a=*/false, /*reuse_b=*/false);
}

static __device__ __forceinline__ unsigned short f2bf(float f) {
    unsigned int u = __float_as_uint(f);
    unsigned int r = u + 0x7FFFu + ((u >> 16) & 1u);   // round-to-nearest-even
    return (unsigned short)(r >> 16);
}

// A-matrix (16x32 bf16) K index for VGPR v, lane half (ISA 7.12.2)
static __device__ __forceinline__ int kmapA(int v, int half) {
    return ((v < 4) ? 2 * v : 2 * v + 8) + 8 * half;   // always even
}
// B-matrix (32x16 bf16) K index for VGPR v, lane half (same order as A)
static __device__ __forceinline__ int kmapB(int v, int half) {
    return ((v < 4) ? 2 * v : 2 * v + 8) + 8 * half;   // always even
}

// ---------------------------------------------------------------------------
// Tensor Data Mover: 2D bf16 tile (tile_d0 x tile_d1 rows, row stride
// stride0 elements) from global -> LDS, packed row-major. D# per ISA 8.3/8.4.
// Issued once per wave; tracked with TENSORcnt.
// ---------------------------------------------------------------------------
static __device__ __forceinline__ void tdm_load_2d_bf16(
    unsigned lds_byte_off, const void* gptr,
    unsigned tensor_d0, unsigned tensor_d1,
    unsigned tile_d0, unsigned tile_d1, unsigned stride0) {
    unsigned long long ga = (unsigned long long)gptr;
    u32x4 g0;
    g0[0] = 1u;                                        // count=1, user mode
    g0[1] = lds_byte_off;                              // lds_addr (bytes)
    g0[2] = (unsigned)ga;                              // global_addr[31:0]
    g0[3] = (unsigned)((ga >> 32) & 0x1FFFFFFu)        // global_addr[56:32]
          | (2u << 30);                                // type = 2 ("image")
    i32x8 g1;
    g1[0] = (int)(1u << 16);                           // data_size=1 (2 bytes)
    g1[1] = (int)(tensor_d0 << 16);                    // tensor_dim0[15:0]
    g1[2] = (int)((tensor_d0 >> 16) | (tensor_d1 << 16));
    g1[3] = (int)((tensor_d1 >> 16) | (tile_d0 << 16));
    g1[4] = (int)(tile_d1 & 0xFFFFu);                  // tile_dim1 (dim2=0)
    g1[5] = (int)stride0;                              // dim0_stride[31:0]
    g1[6] = 0;
    g1[7] = 0;
    i32x4 z4 = {0, 0, 0, 0};
#if defined(__clang_major__) && (__clang_major__ >= 23)
    i32x8 z8 = {0, 0, 0, 0, 0, 0, 0, 0};
    __builtin_amdgcn_tensor_load_to_lds(g0, g1, z4, z4, z8, 0);
#else
    __builtin_amdgcn_tensor_load_to_lds(g0, g1, z4, z4, 0);
#endif
}

static __device__ __forceinline__ unsigned lds_off(const void* p) {
    // LDS aperture maps addr[31:0] directly to the LDS byte offset.
    return (unsigned)(unsigned long long)p;
}

// ---------------------------------------------------------------------------
// fp32 -> bf16 pair conversion
// ---------------------------------------------------------------------------
__global__ void cvt_f32_bf16(const float* __restrict__ src,
                             unsigned int* __restrict__ dst, int npairs) {
    int i = blockIdx.x * blockDim.x + threadIdx.x;
    if (i < npairs) {
        float2 f = ((const float2*)src)[i];
        dst[i] = (unsigned int)f2bf(f.x) | ((unsigned int)f2bf(f.y) << 16);
    }
}

// ---------------------------------------------------------------------------
// GEMM: C[M,1024] = A[M,1024](bf16) * B[1024,1024](bf16) + bias, M = 8192.
// Block tile 128x64, 8 waves of 32x32 (2x2 WMMA tiles), K-step 32.
// Tiles staged to LDS by TDM (wave 0), consumed by all waves.
// ---------------------------------------------------------------------------
template <bool BF16_OUT>
__global__ void __launch_bounds__(256)
gemm_bias(const unsigned int* __restrict__ A,   // bf16 pairs, ROWS x 512
          const unsigned int* __restrict__ B,   // bf16 pairs, 1024 x 512
          const float* __restrict__ bias,
          void* __restrict__ out) {
    __shared__ unsigned int Alds[128 * 16];     // 128 rows x 32 bf16
    __shared__ unsigned int Blds[32 * 32];      // 32  rows x 64 bf16

    const int tid  = threadIdx.x;
    const int lane = tid & 31;
    const int wave = tid >> 5;
    const int half = lane >> 4;
    const int l16  = lane & 15;
    const int waveM = wave & 3;                 // 0..3  -> 32-row strip
    const int waveN = wave >> 2;                // 0..1  -> 32-col strip
    const int rowBase = blockIdx.x * 128;
    const int colBase = blockIdx.y * 64;

    const unsigned aOff = lds_off(Alds);
    const unsigned bOff = lds_off(Blds);

    v8f c[2][2] = {};

    for (int kt = 0; kt < EMBED; kt += 32) {
        __syncthreads();                        // previous tile fully consumed
        if (wave == 0) {
            // A tile: 128 rows x 32 cols @ (rowBase, kt)
            tdm_load_2d_bf16(aOff,
                (const char*)A + ((size_t)rowBase * EMBED + kt) * 2,
                EMBED, ROWS, 32, 128, EMBED);
            // B tile: 32 rows x 64 cols @ (kt, colBase)
            tdm_load_2d_bf16(bOff,
                (const char*)B + ((size_t)kt * EMBED + colBase) * 2,
                EMBED, EMBED, 64, 32, EMBED);
            __builtin_amdgcn_s_wait_tensorcnt(0);
        }
        __syncthreads();                        // tiles visible to all waves

        BF16x16 a[2], b[2];
#pragma unroll
        for (int sm = 0; sm < 2; ++sm) {
            int arow = waveM * 32 + sm * 16 + l16;
#pragma unroll
            for (int v = 0; v < 8; ++v)
                a[sm].u[v] = Alds[arow * 16 + (kmapA(v, half) >> 1)];
        }
        const unsigned short* Bs = (const unsigned short*)Blds;
#pragma unroll
        for (int sn = 0; sn < 2; ++sn) {
            int bcol = waveN * 32 + sn * 16 + l16;
#pragma unroll
            for (int v = 0; v < 8; ++v) {
                int k = kmapB(v, half);
                unsigned short lo = Bs[(k)     * 64 + bcol];
                unsigned short hi = Bs[(k + 1) * 64 + bcol];
                b[sn].u[v] = (unsigned int)lo | ((unsigned int)hi << 16);
            }
        }
#pragma unroll
        for (int sm = 0; sm < 2; ++sm)
#pragma unroll
            for (int sn = 0; sn < 2; ++sn)
                c[sm][sn] = wmma_bf16(a[sm].v, b[sn].v, c[sm][sn]);
    }

    // epilogue: bias add + store (D layout: VGPR r -> row r + 8*half)
    const int trow = rowBase + waveM * 32;
    const int tcol = colBase + waveN * 32;
#pragma unroll
    for (int sm = 0; sm < 2; ++sm)
#pragma unroll
        for (int sn = 0; sn < 2; ++sn) {
            int col = tcol + sn * 16 + l16;
            float bb = bias[col];
#pragma unroll
            for (int r = 0; r < 8; ++r) {
                int row = trow + sm * 16 + r + 8 * half;
                float val = c[sm][sn][r] + bb;
                if (BF16_OUT)
                    ((unsigned short*)out)[(size_t)row * EMBED + col] = f2bf(val);
                else
                    ((float*)out)[(size_t)row * EMBED + col] = val;
            }
        }
}

// ---------------------------------------------------------------------------
// Flash-style attention. One block = (n, h, 128 queries); one wave = 16
// queries. Streams 32 keys/iter with online softmax; fp32 accumulators.
// K/V tiles (32 keys x 64 d, shared by all 8 waves) staged to LDS by TDM.
// ---------------------------------------------------------------------------
__global__ void __launch_bounds__(256)
attention(const unsigned int* __restrict__ Qb,    // bf16 pairs
          const unsigned int* __restrict__ Kb,
          const unsigned int* __restrict__ Vb,
          unsigned short* __restrict__ Ctx) {     // bf16 out
    __shared__ unsigned int Klds[32 * 32];        // 32 keys x 64 d bf16
    __shared__ unsigned int Vlds[32 * 32];        // 32 keys x 64 d bf16
    __shared__ unsigned int Plds[8 * 256];        // per-wave 16x32 bf16 tile

    const int tid  = threadIdx.x;
    const int lane = tid & 31;
    const int wave = tid >> 5;
    const int half = lane >> 4;
    const int l16  = lane & 15;
    const int n    = blockIdx.z;
    const int h    = blockIdx.y;
    const int q0   = blockIdx.x * 128 + wave * 16;
    const size_t rowOff = (size_t)n * SEQ;        // first global row of batch

    const unsigned kOff = lds_off(Klds);
    const unsigned vOff = lds_off(Vlds);

    // Q fragments (two 32-wide d chunks), direct global loads (per-wave data)
    BF16x16 qa[2];
#pragma unroll
    for (int cch = 0; cch < 2; ++cch)
#pragma unroll
        for (int v = 0; v < 8; ++v) {
            int k = kmapA(v, half) + 32 * cch;
            size_t idx = (rowOff + q0 + l16) * EMBED + h * HEAD_DIM + k;
            qa[cch].u[v] = Qb[idx >> 1];
        }

    float m[8], lsum[8];
    v8f acc[4] = {};
#pragma unroll
    for (int r = 0; r < 8; ++r) { m[r] = -INFINITY; lsum[r] = 0.f; }

    unsigned int* pw = &Plds[wave * 256];
    unsigned short* ps = (unsigned short*)pw;

    for (int kt = 0; kt < SEQ; kt += 32) {
        // ---- TDM: stage K/V tiles (32 keys x 64 d) for whole block ----
        if (wave == 0) {
            const size_t eoff = ((rowOff + kt) * EMBED + h * HEAD_DIM) * 2;
            tdm_load_2d_bf16(kOff, (const char*)Kb + eoff,
                             EMBED, ROWS, 64, 32, EMBED);
            tdm_load_2d_bf16(vOff, (const char*)Vb + eoff,
                             EMBED, ROWS, 64, 32, EMBED);
            __builtin_amdgcn_s_wait_tensorcnt(0);
        }
        __syncthreads();

        // ---- S = Q * K^T for 32 keys (two 16-key N tiles) ----
        v8f S[2] = {};
#pragma unroll
        for (int s = 0; s < 2; ++s) {
            int key = s * 16 + l16;               // tile-local key index
#pragma unroll
            for (int cch = 0; cch < 2; ++cch) {
                BF16x16 kb;
#pragma unroll
                for (int v = 0; v < 8; ++v) {
                    int d = kmapB(v, half) + 32 * cch;
                    kb.u[v] = Klds[key * 32 + (d >> 1)];
                }
                S[s] = wmma_bf16(qa[cch].v, kb.v, S[s]);
            }
        }

        // ---- online softmax (rows live in 16-lane groups) ----
        float t[8];
#pragma unroll
        for (int r = 0; r < 8; ++r) {
            float a0 = S[0][r] * 0.125f;          // 1/sqrt(64)
            float a1 = S[1][r] * 0.125f;
            S[0][r] = a0; S[1][r] = a1;
            t[r] = fmaxf(a0, a1);
        }
#pragma unroll
        for (int off = 1; off < 16; off <<= 1)
#pragma unroll
            for (int r = 0; r < 8; ++r)
                t[r] = fmaxf(t[r], __shfl_xor(t[r], off, 16));

        float scale[8];
#pragma unroll
        for (int r = 0; r < 8; ++r) {
            float mn = fmaxf(m[r], t[r]);
            scale[r] = __expf(m[r] - mn);
            m[r] = mn;
        }
        float p0[8], p1[8], rs[8];
#pragma unroll
        for (int r = 0; r < 8; ++r) {
            p0[r] = __expf(S[0][r] - m[r]);
            p1[r] = __expf(S[1][r] - m[r]);
            rs[r] = p0[r] + p1[r];
        }
#pragma unroll
        for (int off = 1; off < 16; off <<= 1)
#pragma unroll
            for (int r = 0; r < 8; ++r)
                rs[r] += __shfl_xor(rs[r], off, 16);
#pragma unroll
        for (int r = 0; r < 8; ++r)
            lsum[r] = lsum[r] * scale[r] + rs[r];
#pragma unroll
        for (int tI = 0; tI < 4; ++tI)
#pragma unroll
            for (int r = 0; r < 8; ++r)
                acc[tI][r] *= scale[r];

        // ---- P: D layout -> A layout via per-wave LDS tile ----
#pragma unroll
        for (int r = 0; r < 8; ++r) {
            ps[(r + 8 * half) * 32 + l16]      = f2bf(p0[r]);
            ps[(r + 8 * half) * 32 + 16 + l16] = f2bf(p1[r]);
        }
        __syncthreads();   // uniform trip count; orders DS store->load
        BF16x16 pa;
#pragma unroll
        for (int v = 0; v < 8; ++v)
            pa.u[v] = pw[l16 * 16 + (kmapA(v, half) >> 1)];

        // ---- acc += P * V (four 16-wide d tiles) ----
        const unsigned short* Vs = (const unsigned short*)Vlds;
#pragma unroll
        for (int tI = 0; tI < 4; ++tI) {
            BF16x16 vb;
            int d = tI * 16 + l16;
#pragma unroll
            for (int v = 0; v < 8; ++v) {
                int kk = kmapB(v, half);
                unsigned short lo = Vs[(kk)     * 64 + d];
                unsigned short hi = Vs[(kk + 1) * 64 + d];
                vb.u[v] = (unsigned int)lo | ((unsigned int)hi << 16);
            }
            acc[tI] = wmma_bf16(pa.v, vb.v, acc[tI]);
        }
        __syncthreads();   // all reads done before next TDM overwrite
    }

    // ---- finalize: ctx = acc / lsum, write bf16 ----
#pragma unroll
    for (int tI = 0; tI < 4; ++tI) {
        int col = tI * 16 + l16;
#pragma unroll
        for (int r = 0; r < 8; ++r) {
            int row = q0 + r + 8 * half;
            float v = acc[tI][r] / lsum[r];
            Ctx[(rowOff + row) * EMBED + h * HEAD_DIM + col] = f2bf(v);
        }
    }
}

// ---------------------------------------------------------------------------
// residual + LayerNorm: one block per row, 256 threads x 4 columns
// ---------------------------------------------------------------------------
__global__ void __launch_bounds__(256)
residual_ln(const float* __restrict__ proj, const float* __restrict__ X0,
            const float* __restrict__ gamma, const float* __restrict__ beta,
            float* __restrict__ out) {
    __shared__ float rS[256], rS2[256];
    const int row = blockIdx.x;
    const int tid = threadIdx.x;
    const float* pr = proj + (size_t)row * EMBED;
    const float* xr = X0   + (size_t)row * EMBED;

    float x[4], s = 0.f, s2 = 0.f;
#pragma unroll
    for (int i = 0; i < 4; ++i) {
        float v = pr[tid + 256 * i] + xr[tid + 256 * i];
        x[i] = v; s += v; s2 += v * v;
    }
    rS[tid] = s; rS2[tid] = s2;
    __syncthreads();
#pragma unroll
    for (int o = 128; o > 0; o >>= 1) {
        if (tid < o) { rS[tid] += rS[tid + o]; rS2[tid] += rS2[tid + o]; }
        __syncthreads();
    }
    float mu  = rS[0]  * (1.0f / EMBED);
    float var = rS2[0] * (1.0f / EMBED) - mu * mu;
    float inv = rsqrtf(var + 1e-8f);
#pragma unroll
    for (int i = 0; i < 4; ++i) {
        int c = tid + 256 * i;
        out[(size_t)row * EMBED + c] = (x[i] - mu) * inv * gamma[c] + beta[c];
    }
}

// ---------------------------------------------------------------------------
// launch
// ---------------------------------------------------------------------------
extern "C" void kernel_launch(void* const* d_in, const int* in_sizes, int n_in,
                              void* d_out, int out_size, void* d_ws, size_t ws_size,
                              hipStream_t stream) {
    const float* X     = (const float*)d_in[0];
    const float* Wq    = (const float*)d_in[1];
    const float* bq    = (const float*)d_in[2];
    const float* Wk    = (const float*)d_in[3];
    const float* bk    = (const float*)d_in[4];
    const float* Wv    = (const float*)d_in[5];
    const float* bv    = (const float*)d_in[6];
    const float* Wp    = (const float*)d_in[7];
    const float* bp    = (const float*)d_in[8];
    const float* gamma = (const float*)d_in[9];
    const float* beta  = (const float*)d_in[10];

    const size_t MiB = 1024u * 1024u;
    char* ws = (char*)d_ws;
    unsigned int* Xb  = (unsigned int*)(ws + 0 * MiB);    // 16 MiB (bf16 X)
    unsigned int* Wqb = (unsigned int*)(ws + 16 * MiB);   // 2 MiB each
    unsigned int* Wkb = (unsigned int*)(ws + 18 * MiB);
    unsigned int* Wvb = (unsigned int*)(ws + 20 * MiB);
    unsigned int* Wpb = (unsigned int*)(ws + 22 * MiB);
    unsigned int* Qb  = (unsigned int*)(ws + 24 * MiB);   // 16 MiB
    unsigned int* Kb  = (unsigned int*)(ws + 40 * MiB);   // 16 MiB
    unsigned int* Vb  = (unsigned int*)(ws + 56 * MiB);   // 16 MiB
    unsigned int* Ctx = (unsigned int*)(ws + 72 * MiB);   // 16 MiB
    float*        Prj = (float*)(ws + 24 * MiB);          // 32 MiB, reuses Q/K

    // 1. fp32 -> bf16
    const int xp = ROWS * EMBED / 2;        // 4,194,304 pairs
    const int wp = EMBED * EMBED / 2;       //   524,288 pairs
    cvt_f32_bf16<<<(xp + 255) / 256, 256, 0, stream>>>(X,  Xb,  xp);
    cvt_f32_bf16<<<(wp + 255) / 256, 256, 0, stream>>>(Wq, Wqb, wp);
    cvt_f32_bf16<<<(wp + 255) / 256, 256, 0, stream>>>(Wk, Wkb, wp);
    cvt_f32_bf16<<<(wp + 255) / 256, 256, 0, stream>>>(Wv, Wvb, wp);
    cvt_f32_bf16<<<(wp + 255) / 256, 256, 0, stream>>>(Wp, Wpb, wp);

    // 2. QKV projections (WMMA GEMM, bf16 out)
    dim3 gGemm(ROWS / 128, EMBED / 64);
    gemm_bias<true><<<gGemm, 256, 0, stream>>>(Xb, Wqb, bq, Qb);
    gemm_bias<true><<<gGemm, 256, 0, stream>>>(Xb, Wkb, bk, Kb);
    gemm_bias<true><<<gGemm, 256, 0, stream>>>(Xb, Wvb, bv, Vb);

    // 3. flash attention
    dim3 gAttn(SEQ / 128, HEADS, BATCH);
    attention<<<gAttn, 256, 0, stream>>>(Qb, Kb, Vb, (unsigned short*)Ctx);

    // 4. output projection (fp32 out)
    gemm_bias<false><<<gGemm, 256, 0, stream>>>(Ctx, Wpb, bp, Prj);

    // 5. residual + LayerNorm
    residual_ln<<<ROWS, 256, 0, stream>>>(Prj, X, gamma, beta, (float*)d_out);
}
